// PNAEmbeddingNet_4612794876595
// MI455X (gfx1250) — matrix operand order repaired and checked by reading, not compiled, not run on hardware
//
#include <hip/hip_runtime.h>
#include <hip/hip_bf16.h>
#include <math.h>

// ---------------- constants (match reference) ----------------
#define NN  32768
#define EE  262144
#define BB  128
#define F0  64
#define DD  128
#define EDD 8
#define LL  6
#define TT  4
#define NEG 0.01f

typedef _Float16 half16 __attribute__((ext_vector_type(16)));
typedef _Float16 half8v __attribute__((ext_vector_type(8)));
typedef float    float8 __attribute__((ext_vector_type(8)));
typedef float    float4v __attribute__((ext_vector_type(4)));

// order-preserving float<->uint for atomic max/min
__device__ __forceinline__ unsigned ordf(float f) {
    unsigned u = __float_as_uint(f);
    return (u & 0x80000000u) ? ~u : (u | 0x80000000u);
}
__device__ __forceinline__ float unordf(unsigned o) {
    unsigned u = (o & 0x80000000u) ? (o & 0x7FFFFFFFu) : ~o;
    return __uint_as_float(u);
}
__device__ __forceinline__ float sigm(float x) { return 1.f / (1.f + expf(-x)); }

// ---------------- generic utility kernels ----------------
__global__ void fill_u32(unsigned* p, unsigned v, size_t n) {
    size_t i = (size_t)blockIdx.x * blockDim.x + threadIdx.x;
    size_t stride = (size_t)gridDim.x * blockDim.x;
    for (; i < n; i += stride) p[i] = v;
}
__global__ void cast_to_f16(const float* __restrict__ s, _Float16* __restrict__ d, size_t n) {
    size_t i = (size_t)blockIdx.x * blockDim.x + threadIdx.x;
    size_t stride = (size_t)gridDim.x * blockDim.x;
    for (; i < n; i += stride) d[i] = (_Float16)s[i];
}

// ---- WMMA GEMM: C[M,Nc] = A[M,K](f16,row-major) * Bt[Nc,K](f16, B transposed) ----
// wave32; 16x64 tile per wave (4 accumulators, 4 v_wmma per 32-K step).
// Fragments use contiguous k order (k = 16*half + s) for A and B. All loads are 16/32-byte vectors.
// outHalf=1 stores the D matrix as f16 (used for Pi/Pj to halve gather traffic).
__global__ __launch_bounds__(256)
void wmma_gemm(const _Float16* __restrict__ A, const _Float16* __restrict__ Bt,
               const float* __restrict__ bias, float* __restrict__ Cm,
               int M, int K, int Nc, int accum, int outHalf) {
    int wave = threadIdx.x >> 5;
    int lane = threadIdx.x & 31;
    int tilesN = Nc >> 6;                    // 64-wide tiles
    int totalTiles = (M >> 4) * tilesN;
    int tile = blockIdx.x * 8 + wave;
    if (tile >= totalTiles) return;
    int tm = tile / tilesN, tn = tile % tilesN;
    int hi = lane >> 4;                      // lane half selects K range / row offset
    int row = (tm << 4) + (lane & 15);       // A row (M) for this lane
    int colBase = (tn << 6) + (lane & 15);   // column for subtile s is colBase + 16*s

    float8 acc[4];
#pragma unroll
    for (int s = 0; s < 4; s++) {
        if (accum) {
#pragma unroll
            for (int r = 0; r < 8; r++)
                acc[s][r] = Cm[(size_t)((tm << 4) + r + hi * 8) * Nc + colBase + s * 16];
        } else {
#pragma unroll
            for (int r = 0; r < 8; r++) acc[s][r] = 0.f;
        }
    }
    const _Float16* Arow = A + (size_t)row * K;
    for (int k0 = 0; k0 < K; k0 += 32) {
        // A 16x32 fragment: lanes 0-15 K 0..15, lanes 16-31 K 16..31
        union { half16 v; half8v h[2]; } a;
        a.h[0] = *(const half8v*)(Arow + k0 + hi * 16);
        a.h[1] = *(const half8v*)(Arow + k0 + hi * 16 + 8);
        int kb = k0 + hi * 16;               // B: lanes 0-15 K 0..15; 16-31 K 16..31
#pragma unroll
        for (int s = 0; s < 4; s++) {
            half16 b = *(const half16*)(Bt + (size_t)(colBase + s * 16) * K + kb);
            acc[s] = __builtin_amdgcn_wmma_f32_16x16x32_f16(false, a.v, false, b,
                                                            (short)0, acc[s], false, false);
        }
    }
    if (outHalf) {
        _Float16* Ch = (_Float16*)Cm;
#pragma unroll
        for (int s = 0; s < 4; s++) {
            float bb = bias ? bias[colBase + s * 16] : 0.f;
#pragma unroll
            for (int r = 0; r < 8; r++)
                Ch[(size_t)((tm << 4) + r + hi * 8) * Nc + colBase + s * 16] =
                    (_Float16)(acc[s][r] + bb);
        }
    } else {
#pragma unroll
        for (int s = 0; s < 4; s++) {
            float bb = bias ? bias[colBase + s * 16] : 0.f;
#pragma unroll
            for (int r = 0; r < 8; r++)
                Cm[(size_t)((tm << 4) + r + hi * 8) * Nc + colBase + s * 16] = acc[s][r] + bb;
        }
    }
}

// ---------------- weight repack kernels (tiny; produce TRANSPOSED B) ----------------
// preW: [T, 3*fin, fin] -> WiT/WjT f16 [C, fin] with C = T*fin
__global__ void pack_pre(const float* __restrict__ preW, _Float16* WiT, _Float16* WjT, int fin) {
    int C = 4 * fin;
    int idx = blockIdx.x * blockDim.x + threadIdx.x;
    if (idx >= C * fin) return;
    int cc = idx / fin, f = idx % fin;
    int t = cc / fin, o = cc % fin;
    size_t base = (size_t)t * 3 * fin * fin;
    WiT[(size_t)cc * fin + f] = (_Float16)preW[base + (size_t)f * fin + o];
    WjT[(size_t)cc * fin + f] = (_Float16)preW[base + (size_t)(fin + f) * fin + o];
}
// fused edge weight Wf[k,C] = sum_f We[f,k]*Wev[t,f,o]; biasE = be@Wev + preb
__global__ void pack_edgefuse(const float* __restrict__ preW, const float* __restrict__ We,
                              const float* __restrict__ be, const float* __restrict__ preb,
                              float* Wf, float* biasE, int fin) {
    int C = 4 * fin;
    int cc = blockIdx.x * blockDim.x + threadIdx.x;
    if (cc >= C) return;
    int t = cc / fin, o = cc % fin;
    size_t base = (size_t)t * 3 * fin * fin + (size_t)2 * fin * fin + o;
    for (int k = 0; k < EDD; k++) {
        float acc = 0.f;
        for (int f = 0; f < fin; f++) acc += We[(size_t)f * EDD + k] * preW[base + (size_t)f * fin];
        Wf[(size_t)k * C + cc] = acc;
    }
    float bev = 0.f;
    for (int f = 0; f < fin; f++) bev += be[f] * preW[base + (size_t)f * fin];
    biasE[cc] = bev + preb[cc];
}
// postW: [T, 4*fin, 32]; x-part dense transposed [128, fin]
__global__ void pack_postx(const float* __restrict__ postW, _Float16* WxT, int fin) {
    int idx = blockIdx.x * blockDim.x + threadIdx.x;
    if (idx >= 128 * fin) return;
    int cc = idx / fin, f = idx % fin;
    int t = cc / 32, o = cc % 32;
    WxT[(size_t)cc * fin + f] = (_Float16)postW[(size_t)t * 4 * fin * 32 + (size_t)f * 32 + o];
}
// block-diagonal tower weights for mean/min/max parts, transposed: [128, C]
__global__ void pack_postbd(const float* __restrict__ postW, _Float16* W, int fin, int part) {
    int C = 4 * fin;
    int idx = blockIdx.x * blockDim.x + threadIdx.x;
    if (idx >= 128 * C) return;
    int cc = idx / C, r = idx % C;
    int tr = r / fin, tc = cc / 32;
    float v = 0.f;
    if (tr == tc)
        v = postW[(size_t)tc * 4 * fin * 32 + (size_t)(part * fin + (r % fin)) * 32 + (cc % 32)];
    W[(size_t)cc * C + r] = (_Float16)v;
}
// lin: x@linW^T -> Bt[c,k] = linW[c,k]: plain f32->f16 cast (done via cast_to_f16)

// -------- fused edge message + segment aggregation --------
// Pi/Pj are f16 [N,C]; 8 features/thread (one b128 gather each).
// Wf (16KB region) + biasE (2KB) staged into LDS once per block via
// global_load_async_to_lds_b128 (ASYNCcnt), then reused from LDS by all threads.
#define WF_BYTES 18432  // 8*512*4 (Wf alloc, max) + 512*4 (biasE), contiguous in ws
__global__ __launch_bounds__(256)
void edge_msg(const _Float16* __restrict__ Pi, const _Float16* __restrict__ Pj,
              const float* __restrict__ WfG, const float* __restrict__ ea,
              const int* __restrict__ ei, float* sum, unsigned* mxo, unsigned* mno,
              float* cnt, int C) {
    __shared__ float sW[WF_BYTES / 4];   // [0..4095]: Wf[k*C+f]; [4096..]: biasE[f]
    // cooperative async stage: each lane moves 16B per issue
    {
        unsigned ldsBase = (unsigned)(size_t)&sW[0];
        const char* gsrc = (const char*)WfG;
        for (int ofs = threadIdx.x * 16; ofs < WF_BYTES; ofs += 256 * 16) {
            unsigned lds = ldsBase + ofs;
            unsigned long long ga = (unsigned long long)(gsrc + ofs);
            asm volatile("global_load_async_to_lds_b128 %0, %1, off"
                         :: "v"(lds), "v"(ga) : "memory");
        }
        asm volatile("s_wait_asynccnt 0x0" ::: "memory");
    }
    __syncthreads();

    int Cq = C >> 3;                       // feature octets per edge
    long long total = (long long)EE * Cq;  // exact multiple of grid*block
    long long stride = (long long)gridDim.x * blockDim.x;
    for (long long idx = (long long)blockIdx.x * blockDim.x + threadIdx.x; idx < total;
         idx += stride) {
        int e = (int)(idx / Cq), fq = (int)(idx % Cq);
        int src = ei[e], dst = ei[EE + e];
        if (fq == 0) atomicAdd(&cnt[dst], 1.f);
        int f0 = fq * 8;
        float m[8];
#pragma unroll
        for (int j = 0; j < 8; j++) m[j] = sW[4096 + f0 + j];  // biasE
#pragma unroll
        for (int k = 0; k < EDD; k++) {
            float av = ea[(size_t)e * EDD + k];
#pragma unroll
            for (int j = 0; j < 8; j++) m[j] += av * sW[k * C + f0 + j];
        }
        union { half8v v; _Float16 s[8]; } pi, pj;
        pi.v = *(const half8v*)(Pi + (size_t)dst * C + f0);
        pj.v = *(const half8v*)(Pj + (size_t)src * C + f0);
        size_t b8 = (size_t)dst * C + f0;
#pragma unroll
        for (int j = 0; j < 8; j++) {
            float mv = m[j] + (float)pi.s[j] + (float)pj.s[j];
            atomicAdd(&sum[b8 + j], mv);
            unsigned om = ordf(mv);
            atomicMax(&mxo[b8 + j], om);
            atomicMin(&mno[b8 + j], om);
        }
    }
}
__global__ void decode_agg(const float* __restrict__ sum, const unsigned* __restrict__ mxo,
                           const unsigned* __restrict__ mno, const float* __restrict__ cnt,
                           _Float16* mean16, _Float16* mn16, _Float16* mx16, int C) {
    size_t idx = (size_t)blockIdx.x * blockDim.x + threadIdx.x;
    if (idx >= (size_t)NN * C) return;
    int n = (int)(idx / C);
    float cn = cnt[n];
    mean16[idx] = (_Float16)(sum[idx] / fmaxf(cn, 1.f));
    mx16[idx]   = (_Float16)(cn > 0.f ? unordf(mxo[idx]) : 0.f);
    mn16[idx]   = (_Float16)(cn > 0.f ? unordf(mno[idx]) : 0.f);
}

// ---------------- batchnorm (training stats) + leaky relu + layer max ----------
__global__ void bn_stats(const float* __restrict__ h, float* mu, float* musq) {
    int f = threadIdx.x;                   // 128 features
    int r0 = blockIdx.x * 64;              // 64 rows per block
    float a = 0.f, b = 0.f;
    for (int r = 0; r < 64; r++) {
        float v = h[(size_t)(r0 + r) * 128 + f];
        a += v; b += v * v;
    }
    atomicAdd(&mu[f], a);
    atomicAdd(&musq[f], b);
}
__global__ void bn_apply(const float* __restrict__ h, const float* __restrict__ mu,
                         const float* __restrict__ musq, const float* __restrict__ gamma,
                         const float* __restrict__ beta, float* hcur, float* hmax, int first) {
    size_t idx = (size_t)blockIdx.x * blockDim.x + threadIdx.x;
    if (idx >= (size_t)NN * 128) return;
    int f = (int)(idx % 128);
    float m = mu[f] * (1.f / NN);
    float var = musq[f] * (1.f / NN) - m * m;
    float v = gamma[f] * (h[idx] - m) * rsqrtf(var + 1e-5f) + beta[f];
    v = (v >= 0.f) ? v : NEG * v;
    hcur[idx] = v;
    hmax[idx] = first ? v : fmaxf(hmax[idx], v);
}

// ---------------- Set2Set ----------------
__global__ __launch_bounds__(512)
void lstm_step(const float* __restrict__ Wih0, const float* __restrict__ Whh0,
               const float* __restrict__ bih0, const float* __restrict__ bhh0,
               const float* __restrict__ Wih1, const float* __restrict__ Whh1,
               const float* __restrict__ bih1, const float* __restrict__ bhh1,
               const float* __restrict__ qs, float* h0, float* c0, float* h1, float* c1) {
    __shared__ float g[512];
    __shared__ float hs[128];
    int b = blockIdx.x, tid = threadIdx.x;
    float acc = bih0[tid] + bhh0[tid];
    for (int j = 0; j < 256; j++) acc += qs[(size_t)b * 256 + j] * Wih0[(size_t)tid * 256 + j];
    for (int j = 0; j < 128; j++) acc += h0[(size_t)b * 128 + j] * Whh0[(size_t)tid * 128 + j];
    g[tid] = acc;
    __syncthreads();
    if (tid < 128) {
        float i = g[tid], f = g[128 + tid], gg = g[256 + tid], o = g[384 + tid];
        float c = sigm(f) * c0[(size_t)b * 128 + tid] + sigm(i) * tanhf(gg);
        float h = sigm(o) * tanhf(c);
        c0[(size_t)b * 128 + tid] = c;
        h0[(size_t)b * 128 + tid] = h;
        hs[tid] = h;
    }
    __syncthreads();
    acc = bih1[tid] + bhh1[tid];
    for (int j = 0; j < 128; j++)
        acc += hs[j] * Wih1[(size_t)tid * 128 + j] + h1[(size_t)b * 128 + j] * Whh1[(size_t)tid * 128 + j];
    g[tid] = acc;
    __syncthreads();
    if (tid < 128) {
        float i = g[tid], f = g[128 + tid], gg = g[256 + tid], o = g[384 + tid];
        float c = sigm(f) * c1[(size_t)b * 128 + tid] + sigm(i) * tanhf(gg);
        float h = sigm(o) * tanhf(c);
        c1[(size_t)b * 128 + tid] = c;
        h1[(size_t)b * 128 + tid] = h;
    }
}
// one wave per node: float4 lane loads + cross-lane reduction (coalesced)
__global__ void attn_dot(const float* __restrict__ hm, const float* __restrict__ q,
                         const int* __restrict__ batch, float* ebuf, unsigned* mo) {
    int wave = threadIdx.x >> 5, lane = threadIdx.x & 31;
    int n = blockIdx.x * 8 + wave;
    if (n >= NN) return;
    int b = batch[n];
    const float4v* hr = (const float4v*)(hm + (size_t)n * 128);
    const float4v* qr = (const float4v*)(q + (size_t)b * 128);
    float4v p = hr[lane] * qr[lane];
    float accv = p[0] + p[1] + p[2] + p[3];
#pragma unroll
    for (int off = 16; off; off >>= 1) accv += __shfl_xor(accv, off, 32);
    if (lane == 0) {
        ebuf[n] = accv;
        atomicMax(&mo[b], ordf(accv));
    }
}
__global__ void attn_exp(const int* __restrict__ batch, const unsigned* __restrict__ mo,
                         float* ebuf, float* sb) {
    int n = blockIdx.x * blockDim.x + threadIdx.x;
    if (n >= NN) return;
    int b = batch[n];
    float m = unordf(mo[b]);
    if (!isfinite(m)) m = 0.f;
    float ex = expf(ebuf[n] - m);
    ebuf[n] = ex;
    atomicAdd(&sb[b], ex);
}
__global__ void attn_r(const float* __restrict__ hm, const int* __restrict__ batch,
                       const float* __restrict__ ebuf, const float* __restrict__ sb, float* rb) {
    size_t idx = (size_t)blockIdx.x * blockDim.x + threadIdx.x;
    if (idx >= (size_t)NN * 128) return;
    int n = (int)(idx / 128), f = (int)(idx % 128);
    int b = batch[n];
    float a = ebuf[n] / fmaxf(sb[b], 1e-16f);
    atomicAdd(&rb[(size_t)b * 128 + f], a * hm[idx]);
}
__global__ void qstar_update(const float* __restrict__ h1, const float* __restrict__ rb, float* qs) {
    int idx = blockIdx.x * blockDim.x + threadIdx.x;
    if (idx >= BB * 256) return;
    int b = idx / 256, f = idx % 256;
    qs[idx] = (f < 128) ? h1[(size_t)b * 128 + f] : rb[(size_t)b * 128 + (f - 128)];
}
__global__ void copy_out(const float* __restrict__ qs, float* out) {
    int idx = blockIdx.x * blockDim.x + threadIdx.x;
    if (idx < BB * 256) out[idx] = qs[idx];
}

// ---------------- orchestration ----------------
extern "C" void kernel_launch(void* const* d_in, const int* in_sizes, int n_in,
                              void* d_out, int out_size, void* d_ws, size_t ws_size,
                              hipStream_t stream) {
    (void)in_sizes; (void)n_in; (void)out_size; (void)ws_size;
    // input flattening: depth-first insertion order of setup_inputs()
    const float* x     = (const float*)d_in[0];
    const int*   ei    = (const int*)d_in[1];
    const float* ea    = (const float*)d_in[2];
    const int*   batch = (const int*)d_in[3];
    auto LP = [&](int l, int j) { return (const float*)d_in[4 + l * 10 + j]; };
    const float* Wih0 = (const float*)d_in[64]; const float* Whh0 = (const float*)d_in[65];
    const float* bih0 = (const float*)d_in[66]; const float* bhh0 = (const float*)d_in[67];
    const float* Wih1 = (const float*)d_in[68]; const float* Whh1 = (const float*)d_in[69];
    const float* bih1 = (const float*)d_in[70]; const float* bhh1 = (const float*)d_in[71];

    char* w = (char*)d_ws;
    size_t off = 0;
    auto alloc = [&](size_t bytes) -> char* {
        off = (off + 255) & ~(size_t)255;
        char* p = w + off;
        off += bytes;
        return p;
    };
    const size_t CM = 512;  // max T*fin
    _Float16* h16    = (_Float16*)alloc((size_t)NN * 128 * 2);
    _Float16* Pi     = (_Float16*)alloc((size_t)NN * CM * 2);   // f16 node projections
    _Float16* Pj     = (_Float16*)alloc((size_t)NN * CM * 2);
    float*    sum    = (float*)alloc((size_t)NN * CM * 4);
    unsigned* mxo    = (unsigned*)alloc((size_t)NN * CM * 4);
    unsigned* mno    = (unsigned*)alloc((size_t)NN * CM * 4);
    float*    cnt    = (float*)alloc((size_t)NN * 4);
    _Float16* mean16 = (_Float16*)alloc((size_t)NN * CM * 2);
    _Float16* mn16   = (_Float16*)alloc((size_t)NN * CM * 2);
    _Float16* mx16   = (_Float16*)alloc((size_t)NN * CM * 2);
    float*    post   = (float*)alloc((size_t)NN * 128 * 4);
    _Float16* post16 = (_Float16*)alloc((size_t)NN * 128 * 2);
    float*    hnew   = (float*)alloc((size_t)NN * 128 * 4);
    float*    hcur   = (float*)alloc((size_t)NN * 128 * 4);
    float*    hmax   = (float*)alloc((size_t)NN * 128 * 4);
    _Float16* WiT    = (_Float16*)alloc(512 * 128 * 2);
    _Float16* WjT    = (_Float16*)alloc(512 * 128 * 2);
    float*    Wf     = (float*)alloc(EDD * 512 * 4);   // 16384 B ...
    float*    biasE  = (float*)alloc(512 * 4);         // ... contiguous: Wf+16384 (WF_BYTES region)
    _Float16* WxT    = (_Float16*)alloc(128 * 128 * 2);
    _Float16* WmT    = (_Float16*)alloc(128 * 512 * 2);
    _Float16* WnT    = (_Float16*)alloc(128 * 512 * 2);
    _Float16* WqT    = (_Float16*)alloc(128 * 512 * 2);
    _Float16* BlT    = (_Float16*)alloc(128 * 128 * 2);
    float*    mu     = (float*)alloc(128 * 4);
    float*    musq   = (float*)alloc(128 * 4);
    float*    ebuf   = (float*)alloc((size_t)NN * 4);
    float*    qs     = (float*)alloc(BB * 256 * 4);
    float*    h0     = (float*)alloc(BB * 128 * 4);
    float*    c0     = (float*)alloc(BB * 128 * 4);
    float*    h1     = (float*)alloc(BB * 128 * 4);
    float*    c1     = (float*)alloc(BB * 128 * 4);
    unsigned* mo     = (unsigned*)alloc(BB * 4);
    float*    sb     = (float*)alloc(BB * 4);
    float*    rb     = (float*)alloc(BB * 128 * 4);

    const unsigned ORD_NEG_INF = 0x007FFFFFu;  // ordf(-inf)
    auto cdiv = [](size_t a, size_t b) { return (unsigned)((a + b - 1) / b); };

    for (int l = 0; l < LL; l++) {
        int fin = (l == 0) ? F0 : DD;
        int C = TT * fin;
        const float* We = LP(l, 0); const float* be = LP(l, 1);
        const float* preW = LP(l, 2); const float* preb = LP(l, 3);
        const float* postW = LP(l, 4); const float* postb = LP(l, 5);
        const float* linW = LP(l, 6); const float* linb = LP(l, 7);
        const float* gamma = LP(l, 8); const float* beta = LP(l, 9);

        // input cast to f16
        cast_to_f16<<<cdiv((size_t)NN * fin, 256), 256, 0, stream>>>(
            l == 0 ? x : hcur, h16, (size_t)NN * fin);
        // pack weights (transposed B layouts)
        pack_pre<<<cdiv((size_t)C * fin, 256), 256, 0, stream>>>(preW, WiT, WjT, fin);
        pack_edgefuse<<<cdiv(C, 256), 256, 0, stream>>>(preW, We, be, preb, Wf, biasE, fin);
        pack_postx<<<cdiv((size_t)128 * fin, 256), 256, 0, stream>>>(postW, WxT, fin);
        pack_postbd<<<cdiv((size_t)128 * C, 256), 256, 0, stream>>>(postW, WmT, fin, 1);
        pack_postbd<<<cdiv((size_t)128 * C, 256), 256, 0, stream>>>(postW, WnT, fin, 2);
        pack_postbd<<<cdiv((size_t)128 * C, 256), 256, 0, stream>>>(postW, WqT, fin, 3);
        cast_to_f16<<<cdiv(128 * 128, 256), 256, 0, stream>>>(linW, BlT, 128 * 128);
        // node projections via WMMA (16x64 wave tiles), f16 output
        unsigned gpi = cdiv((size_t)(NN / 16) * (C / 64), 8);
        wmma_gemm<<<gpi, 256, 0, stream>>>(h16, WiT, nullptr, (float*)Pi, NN, fin, C, 0, 1);
        wmma_gemm<<<gpi, 256, 0, stream>>>(h16, WjT, nullptr, (float*)Pj, NN, fin, C, 0, 1);
        // aggregation buffers
        fill_u32<<<1024, 256, 0, stream>>>((unsigned*)sum, 0u, (size_t)NN * C);
        fill_u32<<<1024, 256, 0, stream>>>(mxo, 0x00000000u, (size_t)NN * C);
        fill_u32<<<1024, 256, 0, stream>>>(mno, 0xFFFFFFFFu, (size_t)NN * C);
        fill_u32<<<128, 256, 0, stream>>>((unsigned*)cnt, 0u, (size_t)NN);
        // fused edge message + scatter (sum / ordered max / ordered min);
        // Wf+biasE staged to LDS via async-to-LDS, grid-stride amortizes staging
        edge_msg<<<8192, 256, 0, stream>>>(Pi, Pj, Wf, ea, ei, sum, mxo, mno, cnt, C);
        decode_agg<<<cdiv((size_t)NN * C, 256), 256, 0, stream>>>(
            sum, mxo, mno, cnt, mean16, mn16, mx16, C);
        // post-MLP as 4 accumulate GEMMs (dense x-part + 3 block-diagonal tower parts)
        unsigned gpo = cdiv((size_t)(NN / 16) * (128 / 64), 8);
        wmma_gemm<<<gpo, 256, 0, stream>>>(h16, WxT, postb, post, NN, fin, 128, 0, 0);
        wmma_gemm<<<gpo, 256, 0, stream>>>(mean16, WmT, nullptr, post, NN, C, 128, 1, 0);
        wmma_gemm<<<gpo, 256, 0, stream>>>(mn16, WnT, nullptr, post, NN, C, 128, 1, 0);
        wmma_gemm<<<gpo, 256, 0, stream>>>(mx16, WqT, nullptr, post, NN, C, 128, 1, 0);
        // lin layer
        cast_to_f16<<<cdiv((size_t)NN * 128, 256), 256, 0, stream>>>(post, post16, (size_t)NN * 128);
        wmma_gemm<<<gpo, 256, 0, stream>>>(post16, BlT, linb, hnew, NN, 128, 128, 0, 0);
        // batchnorm + leaky relu + running max over layers
        fill_u32<<<1, 256, 0, stream>>>((unsigned*)mu, 0u, 128);
        fill_u32<<<1, 256, 0, stream>>>((unsigned*)musq, 0u, 128);
        bn_stats<<<NN / 64, 128, 0, stream>>>(hnew, mu, musq);
        bn_apply<<<cdiv((size_t)NN * 128, 256), 256, 0, stream>>>(
            hnew, mu, musq, gamma, beta, hcur, hmax, l == 0 ? 1 : 0);
    }

    // ---- Set2Set ----
    fill_u32<<<128, 256, 0, stream>>>((unsigned*)qs, 0u, BB * 256);
    fill_u32<<<64, 256, 0, stream>>>((unsigned*)h0, 0u, BB * 128);
    fill_u32<<<64, 256, 0, stream>>>((unsigned*)c0, 0u, BB * 128);
    fill_u32<<<64, 256, 0, stream>>>((unsigned*)h1, 0u, BB * 128);
    fill_u32<<<64, 256, 0, stream>>>((unsigned*)c1, 0u, BB * 128);
    for (int step = 0; step < 5; step++) {
        lstm_step<<<BB, 512, 0, stream>>>(Wih0, Whh0, bih0, bhh0, Wih1, Whh1, bih1, bhh1,
                                          qs, h0, c0, h1, c1);
        fill_u32<<<1, 128, 0, stream>>>(mo, ORD_NEG_INF, BB);
        fill_u32<<<1, 128, 0, stream>>>((unsigned*)sb, 0u, BB);
        fill_u32<<<64, 256, 0, stream>>>((unsigned*)rb, 0u, BB * 128);
        attn_dot<<<NN / 8, 256, 0, stream>>>(hmax, h1, batch, ebuf, mo);
        attn_exp<<<NN / 256, 256, 0, stream>>>(batch, mo, ebuf, sb);
        attn_r<<<cdiv((size_t)NN * 128, 256), 256, 0, stream>>>(hmax, batch, ebuf, sb, rb);
        qstar_update<<<cdiv(BB * 256, 256), 256, 0, stream>>>(h1, rb, qs);
    }
    copy_out<<<cdiv(BB * 256, 256), 256, 0, stream>>>(qs, (float*)d_out);
}
